// Model_40673340293448
// MI455X (gfx1250) — matrix operand order repaired and checked by reading, not compiled, not run on hardware
//
#include <hip/hip_runtime.h>

// ---------------------------------------------------------------------------
// 2-layer LSTM (T=512, B=256, H=64) + mean-over-time + classifier (16 classes)
// Batch rows are independent -> 16 workgroups, each runs the full recurrence
// for one 16-row batch tile using v_wmma_f32_16x16x32_bf16 (f32 accumulate,
// f32 cell state in registers, bf16 activations exchanged through LDS).
//
// v3: gate math uses raw v_exp_f32 / v_rcp_f32 (no IEEE divide expansion),
// biases folded into the exp2 FMA constants so WMMA accumulators start at
// inline zero (no bias-splat mov chains). Weights stay pinned in VGPRs,
// embedding gather stays software-pipelined one step ahead.
// ---------------------------------------------------------------------------

typedef __attribute__((ext_vector_type(16))) __bf16 v16bf;
typedef __attribute__((ext_vector_type(8)))  __bf16 v8bf;
typedef __attribute__((ext_vector_type(8)))  float  v8f;

#define T_STEPS 512
#define BATCH   256
#define HID     64
#define NCLS    16

#define LOG2E  1.4426950408889634f
#define LOG2E2 2.8853901617779808f  // 2*log2(e)

__device__ __forceinline__ v8f wmma_bf16(v16bf a, v16bf b, v8f c) {
  // (neg_a, A, neg_b, B, c_mod, C, reuse_a, reuse_b)
  return __builtin_amdgcn_wmma_f32_16x16x32_bf16(false, a, false, b, (short)0, c,
                                                 false, false);
}

__device__ __forceinline__ v16bf combine8(v8bf lo, v8bf hi) {
  return __builtin_shufflevector(lo, hi, 0, 1, 2, 3, 4, 5, 6, 7,
                                         8, 9, 10, 11, 12, 13, 14, 15);
}

// A-fragment (16x32 bf16) from an LDS bf16 row: elements 0..7 <- K base+0..7,
// elements 8..15 <- K base+16..23  (base already includes kblk*32 + hb*8).
__device__ __forceinline__ v16bf a_from_lds(const __bf16* p) {
  v8bf lo = *(const v8bf*)p;
  v8bf hi = *(const v8bf*)(p + 16);
  return combine8(lo, hi);
}

// Convert two raw f32 octets (already loaded) into an A fragment.
__device__ __forceinline__ v16bf a_from_v8f(v8f lo, v8f hi) {
  v16bf r;
#pragma unroll
  for (int i = 0; i < 8; ++i) {
    r[i]     = (__bf16)lo[i];
    r[i + 8] = (__bf16)hi[i];
  }
  return r;
}

// Fast gate nonlinearities: fma -> v_exp_f32 -> add -> v_rcp_f32.
// sigmoid(z + b)  with nbs = -LOG2E  * b precomputed
__device__ __forceinline__ float sig_b(float z, float nbs) {
  float e = __builtin_amdgcn_exp2f(__builtin_fmaf(z, -LOG2E, nbs));
  return __builtin_amdgcn_rcpf(1.0f + e);
}
// tanh(z + b)     with nbt = -LOG2E2 * b precomputed
__device__ __forceinline__ float tanh_b(float z, float nbt) {
  float e = __builtin_amdgcn_exp2f(__builtin_fmaf(z, -LOG2E2, nbt));
  return __builtin_fmaf(2.0f, __builtin_amdgcn_rcpf(1.0f + e), -1.0f);
}
// tanh(z)
__device__ __forceinline__ float tanh_nb(float z) {
  float e = __builtin_amdgcn_exp2f(z * -LOG2E2);
  return __builtin_fmaf(2.0f, __builtin_amdgcn_rcpf(1.0f + e), -1.0f);
}

// ---------------------------------------------------------------------------
// Pack W0 (128x256) and W1 (128x256) into bf16 WMMA B-fragment order:
//   P[tile][kblk][lane][e] = W[K = kblk*32 + (lane>>4)*8 + (e&7) + (e>>3)*16][N = tile*16 + (lane&15)]
// so a wave loads one B fragment as 32 contiguous bytes per lane.
// ---------------------------------------------------------------------------
__global__ void pack_weights(const float* __restrict__ W0,
                             const float* __restrict__ W1,
                             __bf16* __restrict__ PB0,
                             __bf16* __restrict__ PB1) {
  int idx   = blockIdx.x * blockDim.x + threadIdx.x;  // 0 .. 65535
  int which = idx >> 15;
  int r     = idx & 32767;
  int e     = r & 15;
  int lane  = (r >> 4) & 31;
  int kb    = (r >> 9) & 3;
  int tile  = (r >> 11) & 15;
  int k     = kb * 32 + ((lane >> 4) << 3) + (e & 7) + ((e >> 3) << 4);
  int n     = tile * 16 + (lane & 15);
  const float* W = which ? W1 : W0;
  __bf16*      P = which ? PB1 : PB0;
  P[r] = (__bf16)W[k * 256 + n];
}

// ---------------------------------------------------------------------------
// Main recurrent kernel: one block per 16-row batch tile, 4 waves per block.
// Wave w owns gate-column block jb = w (columns jb*16 .. jb*16+15) and its
// four gate N-tiles {jb, jb+4, jb+8, jb+12} = (g1, g2, g3, o): all four gate
// pre-activations for a given (row, j) land in the same lane / VGPR slot.
// ---------------------------------------------------------------------------
__global__ void __launch_bounds__(128) lstm_kernel(
    const int*   __restrict__ x,    // [T, B] token ids
    const float* __restrict__ emb,  // [NWORD, 64]
    const float* __restrict__ b0,   // [256]
    const float* __restrict__ b1,   // [256]
    const __bf16* __restrict__ PB0, // packed W0
    const __bf16* __restrict__ PB1, // packed W1
    const float* __restrict__ Wc,   // [64, 16]
    const float* __restrict__ bc,   // [16]
    float*       __restrict__ out)  // [256, 16]
{
  // a1[row][0:64]  = current h0 (layer-0 output y0)
  // a1[row][64:128]= current h1 (layer-1 output y1)
  __shared__ __align__(16) __bf16 a1[16 * 128];      // 4 KB
  __shared__ __align__(16) float  meanbuf[16 * 64];  // 4 KB

  const int tid   = threadIdx.x;
  const int lane  = tid & 31;
  const int jb    = tid >> 5;      // wave id == gate column block (0..3)
  const int hb    = lane >> 4;     // lane half
  const int lm    = lane & 15;
  const int mtile = blockIdx.x;    // 0..15 -> batch rows mtile*16..+15
  const int col   = jb * 16 + lm;  // j index (0..63) this lane handles
  const int r_glb = mtile * 16 + lm;

  for (int i = tid; i < 16 * 128; i += 128) a1[i] = (__bf16)0.0f;

  // Biases folded into the exp2 FMA constants (gates q=0..2 are sigmoids,
  // q=3 is the tanh'd candidate).
  float sb0[3], sb1[3], tb0, tb1;
#pragma unroll
  for (int q = 0; q < 3; ++q) {
    sb0[q] = -LOG2E * b0[(jb + q * 4) * 16 + lm];
    sb1[q] = -LOG2E * b1[(jb + q * 4) * 16 + lm];
  }
  tb0 = -LOG2E2 * b0[(jb + 12) * 16 + lm];
  tb1 = -LOG2E2 * b1[(jb + 12) * 16 + lm];

  // ---- pin all weight B-fragments this wave ever needs in VGPRs ----------
  // wb0[q][kb] / wb1[q][kb]: gate tile (jb + 4q), K-block kb.  32 x 8 VGPRs.
  v16bf wb0[4][4], wb1[4][4];
#pragma unroll
  for (int q = 0; q < 4; ++q) {
#pragma unroll
    for (int kb = 0; kb < 4; ++kb) {
      int tile = jb + q * 4;
      wb0[q][kb] = *(const v16bf*)(PB0 + ((tile * 4 + kb) * 32 + lane) * 16);
      wb1[q][kb] = *(const v16bf*)(PB1 + ((tile * 4 + kb) * 32 + lane) * 16);
    }
  }

  // ---- software-pipelined embedding gather for the x_t half of layer 0 ---
  // Raw f32 kept in registers; converted to bf16 only at consumption, so the
  // load-wait lands a full timestep after issue.
  v8f pf_lo0, pf_hi0, pf_lo1, pf_hi1;
  {
    int token = x[r_glb];  // t = 0
    const float* er = emb + token * HID;
    pf_lo0 = *(const v8f*)(er + hb * 8);
    pf_hi0 = *(const v8f*)(er + hb * 8 + 16);
    pf_lo1 = *(const v8f*)(er + 32 + hb * 8);
    pf_hi1 = *(const v8f*)(er + 32 + hb * 8 + 16);
  }

  v8f c0 = {}, c1 = {}, acc = {};
  __syncthreads();

  for (int t = 0; t < T_STEPS; ++t) {
    // ---------------- layer 0: z0 = [x_t, h0] @ W0  (bias in gate FMA) ----
    // Issue LDS loads for the h0 half first so ds latency hides behind the
    // register-resident kb0/kb1 WMMAs.
    v16bf ah2 = a_from_lds(a1 + lm * 128 + 0 * 32 + hb * 8);   // K 64..95
    v16bf ah3 = a_from_lds(a1 + lm * 128 + 1 * 32 + hb * 8);   // K 96..127

    v16bf ax0 = a_from_v8f(pf_lo0, pf_hi0);                    // K 0..31
    v16bf ax1 = a_from_v8f(pf_lo1, pf_hi1);                    // K 32..63

    v8f z[4] = {{}, {}, {}, {}};  // accumulators start at inline zero
#pragma unroll
    for (int q = 0; q < 4; ++q) z[q] = wmma_bf16(ax0, wb0[q][0], z[q]);
#pragma unroll
    for (int q = 0; q < 4; ++q) z[q] = wmma_bf16(ax1, wb0[q][1], z[q]);
#pragma unroll
    for (int q = 0; q < 4; ++q) z[q] = wmma_bf16(ah2, wb0[q][2], z[q]);
#pragma unroll
    for (int q = 0; q < 4; ++q) z[q] = wmma_bf16(ah3, wb0[q][3], z[q]);

    // Prefetch next step's embedding (independent of the recurrence): the
    // token->emb load chain overlaps gates, barriers and the layer-1 GEMM.
    if (t + 1 < T_STEPS) {
      int token = x[(t + 1) * BATCH + r_glb];
      const float* er = emb + token * HID;
      pf_lo0 = *(const v8f*)(er + hb * 8);
      pf_hi0 = *(const v8f*)(er + hb * 8 + 16);
      pf_lo1 = *(const v8f*)(er + 32 + hb * 8);
      pf_hi1 = *(const v8f*)(er + 32 + hb * 8 + 16);
    }

    v8f y0;
#pragma unroll
    for (int v = 0; v < 8; ++v) {
      float s1 = sig_b(z[0][v], sb0[0]);
      float s2 = sig_b(z[1][v], sb0[1]);
      float s3 = sig_b(z[2][v], sb0[2]);
      float to = tanh_b(z[3][v], tb0);
      float cn = c0[v] * s1 + to * s2;
      c0[v]    = cn;
      y0[v]    = tanh_nb(cn) * s3;
    }
    __syncthreads();  // everyone done reading h0 from a1
#pragma unroll
    for (int v = 0; v < 8; ++v)
      a1[(hb * 8 + v) * 128 + col] = (__bf16)y0[v];
    __syncthreads();  // new y0 visible

    // ---------------- layer 1: z1 = [y0, h1] @ W1  (bias in gate FMA) -----
    v16bf aa0 = a_from_lds(a1 + lm * 128 + 0 * 32 + hb * 8);
    v16bf aa1 = a_from_lds(a1 + lm * 128 + 1 * 32 + hb * 8);
    v16bf aa2 = a_from_lds(a1 + lm * 128 + 2 * 32 + hb * 8);
    v16bf aa3 = a_from_lds(a1 + lm * 128 + 3 * 32 + hb * 8);

    v8f w[4] = {{}, {}, {}, {}};
#pragma unroll
    for (int q = 0; q < 4; ++q) w[q] = wmma_bf16(aa0, wb1[q][0], w[q]);
#pragma unroll
    for (int q = 0; q < 4; ++q) w[q] = wmma_bf16(aa1, wb1[q][1], w[q]);
#pragma unroll
    for (int q = 0; q < 4; ++q) w[q] = wmma_bf16(aa2, wb1[q][2], w[q]);
#pragma unroll
    for (int q = 0; q < 4; ++q) w[q] = wmma_bf16(aa3, wb1[q][3], w[q]);

    v8f y1;
#pragma unroll
    for (int v = 0; v < 8; ++v) {
      float s1 = sig_b(w[0][v], sb1[0]);
      float s2 = sig_b(w[1][v], sb1[1]);
      float s3 = sig_b(w[2][v], sb1[2]);
      float to = tanh_b(w[3][v], tb1);
      float cn = c1[v] * s1 + to * s2;
      c1[v]    = cn;
      y1[v]    = tanh_nb(cn) * s3;
      acc[v]  += y1[v];
    }
    __syncthreads();  // everyone done reading h1 from a1
#pragma unroll
    for (int v = 0; v < 8; ++v)
      a1[(hb * 8 + v) * 128 + 64 + col] = (__bf16)y1[v];
    // y1 stores become visible at the first barrier of the next iteration,
    // which precedes the next layer-1 GEMM read of a1[:,64:128].
  }

  // ---------------- mean over time + classifier (f32) ---------------------
  __syncthreads();
#pragma unroll
  for (int v = 0; v < 8; ++v)
    meanbuf[(hb * 8 + v) * HID + col] = acc[v] * (1.0f / (float)T_STEPS);
  __syncthreads();

  for (int o = tid; o < 16 * NCLS; o += 128) {
    int rr = o >> 4, cc = o & 15;
    float s = bc[cc];
#pragma unroll 4
    for (int j = 0; j < HID; ++j) s += meanbuf[rr * HID + j] * Wc[j * NCLS + cc];
    out[(mtile * 16 + rr) * NCLS + cc] = s;
  }
}

// ---------------------------------------------------------------------------
extern "C" void kernel_launch(void* const* d_in, const int* in_sizes, int n_in,
                              void* d_out, int out_size, void* d_ws, size_t ws_size,
                              hipStream_t stream) {
  (void)in_sizes; (void)n_in; (void)out_size; (void)ws_size;
  const int*   x   = (const int*)  d_in[0];
  const float* emb = (const float*)d_in[1];
  const float* W0  = (const float*)d_in[2];
  const float* b0  = (const float*)d_in[3];
  const float* W1  = (const float*)d_in[4];
  const float* b1  = (const float*)d_in[5];
  const float* Wc  = (const float*)d_in[6];
  const float* bc  = (const float*)d_in[7];
  float* out = (float*)d_out;

  __bf16* PB0 = (__bf16*)d_ws;                 // 16*4*32*16 bf16 = 64 KB
  __bf16* PB1 = PB0 + 16 * 4 * 32 * 16;        // another 64 KB

  pack_weights<<<256, 256, 0, stream>>>(W0, W1, PB0, PB1);
  lstm_kernel<<<16, 128, 0, stream>>>(x, emb, b0, b1, PB0, PB1, Wc, bc, out);
}
